// BinaryTreeGRU_10179072491853
// MI455X (gfx1250) — matrix operand order repaired and checked by reading, not compiled, not run on hardware
//
#include <hip/hip_runtime.h>

typedef __attribute__((ext_vector_type(16))) _Float16 v16h;
typedef __attribute__((ext_vector_type(8)))  float    v8f;

#define NODES_OUT 1023   // 2*512 - 1 nodes per batch element in output

__device__ __forceinline__ float sigm(float x) { return 1.0f / (1.0f + __expf(-x)); }

// B fragment (32x16, K x N) from f16 row-major weights W[O][C]:
// lane l: column N = o0 + (l&15); K = k0 + (l>>4)*8 .. +7, then k0 + 16 + (l>>4)*8 .. +7.
__device__ __forceinline__ v16h load_b_frag(const _Float16* __restrict__ W, int ldc,
                                            int o0, int k0, int lane) {
  const _Float16* p = W + (size_t)(o0 + (lane & 15)) * ldc + k0 + ((lane >> 4) << 3);
  v16h b;   // two 16B-aligned 8-half chunks
#pragma unroll
  for (int i = 0; i < 8; ++i) { b[i] = p[i]; b[8 + i] = p[16 + i]; }
  return b;
}

// A fragment (16x32) from an f32 tile in LDS (row-major, leading dim ld), cvt on the fly.
__device__ __forceinline__ v16h frag_a_lds_f32(const float* Af, int ld, int k0, int lane) {
  const float* p = Af + (lane & 15) * ld + k0 + ((lane >> 4) << 3);
  v16h a;
#pragma unroll
  for (int i = 0; i < 8; ++i) a[i] = (_Float16)p[i];
#pragma unroll
  for (int i = 0; i < 8; ++i) a[8 + i] = (_Float16)p[16 + i];
  return a;
}

// CDNA5 async copy: 16B global -> LDS, tracked by ASYNCcnt.
__device__ __forceinline__ void async_g2l_b128(unsigned lds_byte_off, const float* gp) {
  asm volatile("global_load_async_to_lds_b128 %0, %1, off"
               :: "v"(lds_byte_off), "v"(gp) : "memory");
}
__device__ __forceinline__ void wait_async0() {
  asm volatile("s_wait_asynccnt 0x0" ::: "memory");
}

__global__ __launch_bounds__(256)
void cvt_weights(const float* __restrict__ Wgrzx, const float* __restrict__ Wrzh,
                 const float* __restrict__ Wgh,
                 _Float16* __restrict__ wg, _Float16* __restrict__ wr,
                 _Float16* __restrict__ wh) {
  int i = blockIdx.x * 256 + threadIdx.x;
  if (i < 768 * 256)  wg[i] = (_Float16)Wgrzx[i];
  if (i < 1024 * 512) wr[i] = (_Float16)Wrzh[i];
  if (i < 256 * 256)  wh[i] = (_Float16)Wgh[i];
}

// ---------------- Level 0: h = (1 - (sig(a1)+sig(a2))/2) * tanh(a0), a = x*Wgrzx^T + b
__global__ __launch_bounds__(256)
void tree_gru_level0(const float* __restrict__ x,
                     const _Float16* __restrict__ Wg16,
                     const float* __restrict__ bias,
                     float* __restrict__ out) {
  // region0: phase A = A-tile f32 [16][256] (16 KB); phase B = grzx f16 [16][768] (24 KB)
  __shared__ __attribute__((aligned(16))) char smem0[16 * 768 * 2];
  float*    Af  = (float*)smem0;
  _Float16* gz  = (_Float16*)smem0;

  const int tid  = threadIdx.x;
  const int lane = tid & 31;
  const int wave = tid >> 5;
  const int tile = blockIdx.x;

  // ---- async-stage A tile (16 x 256 f32) into LDS
  {
    unsigned lbase = (unsigned)(size_t)(void*)smem0;
#pragma unroll
    for (int j = 0; j < 4; ++j) {
      int idx4 = (tid + j * 256) * 4;
      int row  = idx4 >> 8;
      int col  = idx4 & 255;
      const float* gp = x + (size_t)(tile * 16 + row) * 256 + col;
      async_g2l_b128(lbase + (unsigned)idx4 * 4u, gp);
    }
    wait_async0();
  }
  __syncthreads();

  // ---- GEMM: grzx[16 x 768] = A[16 x 256] * Wgrzx^T ; pipelined 3+3 tiles/wave
  v8f acc[6] = {};
  {
    v16h bcur[3];
#pragma unroll
    for (int t = 0; t < 3; ++t) bcur[t] = load_b_frag(Wg16, 256, wave * 96 + t * 16, 0, lane);
    v16h a = frag_a_lds_f32(Af, 256, 0, lane);
    for (int k0 = 0; k0 < 256; k0 += 32) {
      int k1 = (k0 + 32) & 255;   // wrap keeps dead prefetch in-bounds
      v16h bnext[3];
#pragma unroll
      for (int t = 0; t < 3; ++t) bnext[t] = load_b_frag(Wg16, 256, wave * 96 + (3 + t) * 16, k0, lane);
#pragma unroll
      for (int t = 0; t < 3; ++t)
        acc[t] = __builtin_amdgcn_wmma_f32_16x16x32_f16(false, a, false, bcur[t],
                                                        (short)0, acc[t], false, false);
#pragma unroll
      for (int t = 0; t < 3; ++t) bcur[t] = load_b_frag(Wg16, 256, wave * 96 + t * 16, k1, lane);
      v16h an = frag_a_lds_f32(Af, 256, k1, lane);
#pragma unroll
      for (int t = 0; t < 3; ++t)
        acc[3 + t] = __builtin_amdgcn_wmma_f32_16x16x32_f16(false, a, false, bnext[t],
                                                            (short)0, acc[3 + t], false, false);
      a = an;
    }
  }
  __syncthreads();   // A region dead; reuse as grzx f16

  {
    const int mrow = (lane >> 4) << 3;
    const int ncol = lane & 15;
#pragma unroll
    for (int t = 0; t < 6; ++t) {
      int n = wave * 96 + t * 16 + ncol;
#pragma unroll
      for (int i = 0; i < 8; ++i)
        gz[(mrow + i) * 768 + n] = (_Float16)acc[t][i];
    }
  }
  __syncthreads();

  const int node = tid >> 4;
  const int mb   = (tid & 15) * 16;
  const int r    = tile * 16 + node;
  const int bb   = r >> 9;
  const int leaf = r & 511;
  float* orow = out + ((size_t)bb * NODES_OUT + leaf) * 256;
#pragma unroll
  for (int i = 0; i < 16; ++i) {
    int m = mb + i;
    float a0 = (float)gz[node * 768 + m]        + bias[m];
    float a1 = (float)gz[node * 768 + 256 + m]  + bias[256 + m];
    float a2 = (float)gz[node * 768 + 512 + m]  + bias[512 + m];
    orow[m] = (1.0f - 0.5f * (sigm(a1) + sigm(a2))) * tanhf(a0);
  }
}

// ---------------- Levels 1..9 (x==0)
__global__ __launch_bounds__(256)
void tree_gru_level(float* __restrict__ out,
                    const _Float16* __restrict__ Wr16,
                    const _Float16* __restrict__ Wh16,
                    const float* __restrict__ bias,
                    int off_prev, int off, int K) {
  // region0: phase A = child f32 [16][512] (32 KB); phase B = rzh f16 [16][1024] (32 KB)
  __shared__ __attribute__((aligned(16))) char smem0[16 * 512 * 4];
  __shared__ _Float16 lds_s[16 * 256];           // 8 KB
  __shared__ _Float16 lds_g[16 * 256];           // 8 KB
  float*    Af  = (float*)smem0;
  _Float16* rzh = (_Float16*)smem0;

  const int tid  = threadIdx.x;
  const int lane = tid & 31;
  const int wave = tid >> 5;
  const int tile = blockIdx.x;

  // ---- async-stage child_h tile (16 x 512 f32) into LDS
  {
    unsigned lbase = (unsigned)(size_t)(void*)smem0;
#pragma unroll
    for (int j = 0; j < 8; ++j) {
      int idx4 = (tid + j * 256) * 4;
      int row  = idx4 >> 9;
      int col  = idx4 & 511;
      int rg = tile * 16 + row;
      int bg = rg / K;
      int kg = rg - bg * K;
      const float* gp = out + ((size_t)bg * NODES_OUT + off_prev + 2 * kg) * 256 + col;
      async_g2l_b128(lbase + (unsigned)idx4 * 4u, gp);
    }
    wait_async0();
  }
  __syncthreads();

  // ---- GEMM1: rzh[16 x 1024] = child[16 x 512] * Wrzh^T ; pipelined 4+4 tiles/wave
  v8f acc[8] = {};
  {
    v16h bcur[4];
#pragma unroll
    for (int t = 0; t < 4; ++t) bcur[t] = load_b_frag(Wr16, 512, wave * 128 + t * 16, 0, lane);
    v16h a = frag_a_lds_f32(Af, 512, 0, lane);
    for (int k0 = 0; k0 < 512; k0 += 32) {
      int k1 = (k0 + 32) & 511;
      v16h bnext[4];
#pragma unroll
      for (int t = 0; t < 4; ++t) bnext[t] = load_b_frag(Wr16, 512, wave * 128 + (4 + t) * 16, k0, lane);
#pragma unroll
      for (int t = 0; t < 4; ++t)
        acc[t] = __builtin_amdgcn_wmma_f32_16x16x32_f16(false, a, false, bcur[t],
                                                        (short)0, acc[t], false, false);
#pragma unroll
      for (int t = 0; t < 4; ++t) bcur[t] = load_b_frag(Wr16, 512, wave * 128 + t * 16, k1, lane);
      v16h an = frag_a_lds_f32(Af, 512, k1, lane);
#pragma unroll
      for (int t = 0; t < 4; ++t)
        acc[4 + t] = __builtin_amdgcn_wmma_f32_16x16x32_f16(false, a, false, bnext[t],
                                                            (short)0, acc[4 + t], false, false);
      a = an;
    }
  }
  __syncthreads();

  // ---- pull this thread's exact-f32 child values into registers before aliasing
  const int node = tid >> 4;
  const int mb   = (tid & 15) * 16;
  float c0v[16], c1v[16];
#pragma unroll
  for (int i = 0; i < 16; ++i) {
    c0v[i] = Af[node * 512 + mb + i];
    c1v[i] = Af[node * 512 + 256 + mb + i];
  }
  __syncthreads();

  // ---- store rzh (f16) into region0
  {
    const int mrow = (lane >> 4) << 3;
    const int ncol = lane & 15;
#pragma unroll
    for (int t = 0; t < 8; ++t) {
      int n = wave * 128 + t * 16 + ncol;
#pragma unroll
      for (int i = 0; i < 8; ++i)
        rzh[(mrow + i) * 1024 + n] = (_Float16)acc[t][i];
    }
  }
  __syncthreads();

  // ---- gates: s = r0*ch0 + r1*ch1 (LDS f16); zc, zs in registers
  float zc[16], zs[16];
#pragma unroll
  for (int i = 0; i < 16; ++i) {
    int m = mb + i;
    float b0 = bias[256 + m], b1 = bias[512 + m];
    float r0 = sigm((float)rzh[node * 1024 + m]        + b0);
    float r1 = sigm((float)rzh[node * 1024 + 256 + m]  + b1);
    float z0 = sigm((float)rzh[node * 1024 + 512 + m]  + b0);
    float z1 = sigm((float)rzh[node * 1024 + 768 + m]  + b1);
    lds_s[node * 256 + m] = (_Float16)(r0 * c0v[i] + r1 * c1v[i]);
    zc[i] = z0 * c0v[i] + z1 * c1v[i];
    zs[i] = z0 + z1;
  }
  __syncthreads();

  // ---- GEMM2: g_pre[16 x 256] = s * Wgh^T ; pipelined 2 tiles/wave
  {
    v8f acc2[2] = {};
    const int abase = (lane & 15) * 256 + ((lane >> 4) << 3);
    v16h a;
#pragma unroll
    for (int i = 0; i < 8; ++i) a[i]     = lds_s[abase + i];
#pragma unroll
    for (int i = 0; i < 8; ++i) a[8 + i] = lds_s[abase + 16 + i];
    v16h b0 = load_b_frag(Wh16, 256, wave * 32,      0, lane);
    v16h b1 = load_b_frag(Wh16, 256, wave * 32 + 16, 0, lane);
    for (int k0 = 0; k0 < 256; k0 += 32) {
      int k1 = (k0 + 32) & 255;
      v16h bn0 = load_b_frag(Wh16, 256, wave * 32,      k1, lane);
      v16h bn1 = load_b_frag(Wh16, 256, wave * 32 + 16, k1, lane);
      v16h an;
#pragma unroll
      for (int i = 0; i < 8; ++i) an[i]     = lds_s[abase + k1 + i];
#pragma unroll
      for (int i = 0; i < 8; ++i) an[8 + i] = lds_s[abase + k1 + 16 + i];
      acc2[0] = __builtin_amdgcn_wmma_f32_16x16x32_f16(false, a, false, b0,
                                                       (short)0, acc2[0], false, false);
      acc2[1] = __builtin_amdgcn_wmma_f32_16x16x32_f16(false, a, false, b1,
                                                       (short)0, acc2[1], false, false);
      a = an; b0 = bn0; b1 = bn1;
    }
    const int mrow = (lane >> 4) << 3;
    const int ncol = lane & 15;
#pragma unroll
    for (int t = 0; t < 2; ++t) {
      int n = wave * 32 + t * 16 + ncol;
#pragma unroll
      for (int i = 0; i < 8; ++i)
        lds_g[(mrow + i) * 256 + n] = (_Float16)(t == 0 ? acc2[0][i] : acc2[1][i]);
    }
  }
  __syncthreads();

  // ---- combine: h = zc + (1 - zs/2) * tanh(b[m] + g_pre)
  const int r  = tile * 16 + node;
  const int bb = r / K;
  const int kk = r - bb * K;
  float* orow = out + ((size_t)bb * NODES_OUT + off + kk) * 256;
#pragma unroll
  for (int i = 0; i < 16; ++i) {
    int m = mb + i;
    float g = tanhf((float)lds_g[node * 256 + m] + bias[m]);
    orow[m] = zc[i] + (1.0f - 0.5f * zs[i]) * g;
  }
}

extern "C" void kernel_launch(void* const* d_in, const int* in_sizes, int n_in,
                              void* d_out, int out_size, void* d_ws, size_t ws_size,
                              hipStream_t stream) {
  (void)in_sizes; (void)n_in; (void)out_size; (void)ws_size;
  const float* inputs = (const float*)d_in[0];
  const float* Wgrzx  = (const float*)d_in[1];
  const float* bgrzx  = (const float*)d_in[2];
  const float* Wrzh   = (const float*)d_in[3];
  const float* Wgh    = (const float*)d_in[4];
  float* out = (float*)d_out;

  _Float16* wg = (_Float16*)d_ws;          // [768 x 256]
  _Float16* wr = wg + 768 * 256;           // [1024 x 512]
  _Float16* wh = wr + 1024 * 512;          // [256 x 256]

  cvt_weights<<<2048, 256, 0, stream>>>(Wgrzx, Wrzh, Wgh, wg, wr, wh);

  tree_gru_level0<<<2048, 256, 0, stream>>>(inputs, wg, bgrzx, out);

  int off_prev = 0;
  for (int lvl = 1; lvl <= 9; ++lvl) {
    int Kprev = 512 >> (lvl - 1);
    int K     = 512 >> lvl;
    int off   = off_prev + Kprev;
    int tiles = (64 * K) / 16;
    tree_gru_level<<<tiles, 256, 0, stream>>>(out, wr, wh, bgrzx, off_prev, off, K);
    off_prev = off;
  }
}
